// MLA_59889023975551
// MI455X (gfx1250) — matrix operand order repaired and checked by reading, not compiled, not run on hardware
//
#include <hip/hip_runtime.h>

// ---------------------------------------------------------------------------
// MLA forward for MI455X (gfx1250, wave32, WMMA f32_16x16x32_f16)
// Pipeline: cast fp32->f16, async-to-LDS double-buffered WMMA GEMMs,
// RMS norms, flash-attention (online softmax) with WMMA for QK^T and PV.
// Mask input is the standard causal tril -> computed analytically (start_pos=0).
// ---------------------------------------------------------------------------

typedef __attribute__((ext_vector_type(16))) _Float16 v16h;
typedef __attribute__((ext_vector_type(8)))  _Float16 v8h;
typedef __attribute__((ext_vector_type(8)))  float    v8f;

#define Bn   2
#define Sn   1024
#define Dn   4096
#define Hn   32
#define DHn  128
#define DVn  128
#define RQn  1536
#define RKVn 512
#define EPSF 1.1920929e-7f
#define SMSCALE 0.08838834764831845f   // 128^-0.5

// ---------------- fragment loaders (row-major, K-contiguous) ---------------

// A-fragment 16x32 f16: lane<16 holds row M=lane, K={k0..k0+7, k0+16..k0+23};
// lane>=16 holds row M=lane-16, K={k0+8..k0+15, k0+24..k0+31}.
__device__ __forceinline__ v16h load_frag_a(const _Float16* base, int ld,
                                            int row0, int k0) {
  int lane = threadIdx.x & 31;
  int r  = lane & 15;
  int hh = lane >> 4;
  const _Float16* p = base + (size_t)(row0 + r) * (size_t)ld + k0 + hh * 8;
  v8h lo = *(const v8h*)(p);
  v8h hi = *(const v8h*)(p + 16);
  return __builtin_shufflevector(lo, hi, 0,1,2,3,4,5,6,7,8,9,10,11,12,13,14,15);
}

// B-fragment 32x16 f16 built from W[N][K] row-major (so B = W^T):
// lane<16 holds col N=col0+lane, K={k0..k0+7, k0+16..k0+23}; lane>=16 holds K={k0+8..k0+15, k0+24..k0+31}.
__device__ __forceinline__ v16h load_frag_b(const _Float16* base, int ld,
                                            int col0, int k0) {
  int lane = threadIdx.x & 31;
  int n  = lane & 15;
  int hh = lane >> 4;
  const _Float16* p = base + (size_t)(col0 + n) * (size_t)ld + k0 + hh * 8;
  v8h lo = *(const v8h*)(p);
  v8h hi = *(const v8h*)(p + 16);
  return __builtin_shufflevector(lo, hi, 0,1,2,3,4,5,6,7,8,9,10,11,12,13,14,15);
}

__device__ __forceinline__ v8f wmma16(v16h a, v16h b, v8f c) {
  return __builtin_amdgcn_wmma_f32_16x16x32_f16(false, a, false, b,
                                                (short)0, c, false, false);
}

// Async copy 16B from global to LDS (per lane); tracked by ASYNCcnt.
__device__ __forceinline__ void async_copy_b128(unsigned lds_off,
                                                const void* gaddr) {
  asm volatile("global_load_async_to_lds_b128 %0, %1, off"
               :: "v"(lds_off), "v"(gaddr) : "memory");
}
__device__ __forceinline__ void wait_async0() {
  asm volatile("s_wait_asynccnt 0" ::: "memory");
}

// ---------------------------- cast kernel ----------------------------------

__global__ void cast_f32_to_f16_kernel(const float* __restrict__ in,
                                       _Float16* __restrict__ out, size_t n) {
  size_t i = (size_t)blockIdx.x * blockDim.x + threadIdx.x;
  if (i < n) out[i] = (_Float16)in[i];
}

// ------------------------------ RMS norm -----------------------------------

__global__ __launch_bounds__(256)
void rmsnorm_kernel(const float* __restrict__ in, const float* __restrict__ w,
                    _Float16* __restrict__ out, int R) {
  int row = blockIdx.x;
  const float* x = in + (size_t)row * R;
  float ss = 0.f;
  for (int i = threadIdx.x; i < R; i += 256) { float v = x[i]; ss += v * v; }
  __shared__ float red[256];
  red[threadIdx.x] = ss;
  __syncthreads();
  for (int s = 128; s > 0; s >>= 1) {
    if ((int)threadIdx.x < s) red[threadIdx.x] += red[threadIdx.x + s];
    __syncthreads();
  }
  float inv = rsqrtf(red[0] / (float)R + EPSF);
  for (int i = threadIdx.x; i < R; i += 256)
    out[(size_t)row * R + i] = (_Float16)(x[i] * inv * w[i]);
}

// ------------------------- WMMA GEMM: C = A * W^T + b ----------------------
// A: [M][K] f16 row-major, W: [N][K] f16 row-major, bias: [N] f32.
// Block: 256 threads = 8 waves (2 along M x 4 along N); block tile 128x128;
// wave tile 64x32 = 4x2 WMMA accumulators. K-slices of 32 are staged into
// LDS with double-buffered async-to-LDS copies (overlap copy with WMMA).

template <bool HALF_OUT>
__global__ __launch_bounds__(256)
void gemm_xwT_kernel(const _Float16* __restrict__ A,
                     const _Float16* __restrict__ W,
                     const float* __restrict__ bias,
                     void* __restrict__ Cout, int M, int N, int K) {
  __shared__ _Float16 lds_a[2][128 * 32];   // [buf][m_local][k_local]
  __shared__ _Float16 lds_w[2][128 * 32];   // [buf][n_local][k_local]

  int tid  = threadIdx.x;
  int w    = tid >> 5;
  int lane = tid & 31;
  int wm = w >> 2, wn = w & 3;
  int mblk = blockIdx.y * 128;
  int nblk = blockIdx.x * 128;

  // Stage one 128x32 A tile + 128x32 W tile into LDS buffer `buf`.
  // 8KB each -> 512 x 16B chunks each; 256 threads x 2 chunks per matrix.
  auto stage = [&](int buf, int k0) {
#pragma unroll
    for (int j = 0; j < 2; ++j) {
      int id  = tid + 256 * j;     // 0..511
      int row = id >> 2;           // 0..127
      int c16 = id & 3;            // 16B chunk within 64B row
      const _Float16* ga = A + (size_t)(mblk + row) * K + k0 + c16 * 8;
      unsigned la = (unsigned)(uintptr_t)&lds_a[buf][row * 32 + c16 * 8];
      async_copy_b128(la, ga);
      const _Float16* gw = W + (size_t)(nblk + row) * K + k0 + c16 * 8;
      unsigned lw = (unsigned)(uintptr_t)&lds_w[buf][row * 32 + c16 * 8];
      async_copy_b128(lw, gw);
    }
  };

  v8f acc[4][2];
#pragma unroll
  for (int i = 0; i < 4; ++i)
#pragma unroll
    for (int j = 0; j < 2; ++j) acc[i][j] = {};

  stage(0, 0);
  wait_async0();
  __syncthreads();

  for (int k0 = 0; k0 < K; k0 += 32) {
    int cur = (k0 >> 5) & 1;
    if (k0 + 32 < K) stage(cur ^ 1, k0 + 32);   // prefetch next slice

    const _Float16* la = lds_a[cur];
    const _Float16* lw = lds_w[cur];
    v16h bf0 = load_frag_b(lw, 32, wn * 32,      0);
    v16h bf1 = load_frag_b(lw, 32, wn * 32 + 16, 0);
#pragma unroll
    for (int ms = 0; ms < 4; ++ms) {
      v16h af = load_frag_a(la, 32, wm * 64 + ms * 16, 0);
      acc[ms][0] = wmma16(af, bf0, acc[ms][0]);
      acc[ms][1] = wmma16(af, bf1, acc[ms][1]);
    }
    wait_async0();       // next buffer resident (this wave's copies)
    __syncthreads();     // all waves done reading cur + all copies landed
  }

  int r = lane & 15, hh = lane >> 4;
#pragma unroll
  for (int ms = 0; ms < 4; ++ms)
#pragma unroll
    for (int ns = 0; ns < 2; ++ns) {
      int n = nblk + wn * 32 + ns * 16 + r;
      float bv = bias[n];
#pragma unroll
      for (int j = 0; j < 8; ++j) {
        int m = mblk + wm * 64 + ms * 16 + hh * 8 + j;
        float val = acc[ms][ns][j] + bv;
        if (HALF_OUT)
          ((_Float16*)Cout)[(size_t)m * N + n] = (_Float16)val;
        else
          ((float*)Cout)[(size_t)m * N + n] = val;
      }
    }
}

// ------------------------- flash attention (WMMA) --------------------------
// One wave per (b, h, 16-query tile). Causal, start_pos == 0.
// Q: [B,S,H*128]; KV: [B,S,H*256] (k: e<128, v: e>=128); O: [B,S,H*128].

__global__ __launch_bounds__(32)
void mla_attention_kernel(const _Float16* __restrict__ Q,
                          const _Float16* __restrict__ KV,
                          _Float16* __restrict__ O) {
  int qt = blockIdx.x, h = blockIdx.y, b = blockIdx.z;
  int qbase = qt * 16;
  int lane = threadIdx.x;
  int r = lane & 15, hh = lane >> 4;

  __shared__ _Float16 lds_p[16 * 32];     // P tile, row-major [m][t_local]
  __shared__ _Float16 lds_vt[128 * 32];   // V^T tile, [dv][t_local]

  const _Float16* qp  = Q  + ((size_t)(b * Sn + qbase)) * (Hn * DHn) + h * DHn;
  const _Float16* kvb = KV + ((size_t)(b * Sn)) * (Hn * 256) + h * 256;

  // Q A-fragments, fixed for whole key loop (DH=128 -> 4 K-slices of 32)
  v16h qf[4];
#pragma unroll
  for (int kc = 0; kc < 4; ++kc) qf[kc] = load_frag_a(qp, Hn * DHn, 0, kc * 32);

  v8f acc[8];
#pragma unroll
  for (int i = 0; i < 8; ++i) acc[i] = {};
  float run_max[8], run_sum[8];
#pragma unroll
  for (int j = 0; j < 8; ++j) { run_max[j] = -__builtin_inff(); run_sum[j] = 0.f; }

  for (int tb = 0; tb < qbase + 16; tb += 32) {
    // ----- scores S = Q K^T for keys [tb, tb+32) -----
    v8f sc[2] = {v8f{}, v8f{}};
#pragma unroll
    for (int nt = 0; nt < 2; ++nt)
#pragma unroll
      for (int kc = 0; kc < 4; ++kc) {
        v16h kf = load_frag_b(kvb, Hn * 256, tb + nt * 16, kc * 32);
        sc[nt] = wmma16(qf[kc], kf, sc[nt]);
      }

    // ----- causal mask + scale + online softmax row stats -----
    float corr[8];
#pragma unroll
    for (int j = 0; j < 8; ++j) {
      int m = j + hh * 8;
      int sg = qbase + m;
      float v0 = (tb + r      <= sg) ? sc[0][j] * SMSCALE : -__builtin_inff();
      float v1 = (tb + 16 + r <= sg) ? sc[1][j] * SMSCALE : -__builtin_inff();
      float mx = fmaxf(v0, v1);
#pragma unroll
      for (int d = 1; d < 16; d <<= 1) mx = fmaxf(mx, __shfl_xor(mx, d, 32));
      float nm = fmaxf(run_max[j], mx);
      corr[j] = __expf(run_max[j] - nm);
      run_max[j] = nm;
      float p0 = __expf(v0 - nm);
      float p1 = __expf(v1 - nm);
      sc[0][j] = p0; sc[1][j] = p1;
      float sm = p0 + p1;
#pragma unroll
      for (int d = 1; d < 16; d <<= 1) sm += __shfl_xor(sm, d, 32);
      run_sum[j] = run_sum[j] * corr[j] + sm;
    }
#pragma unroll
    for (int i = 0; i < 8; ++i)
#pragma unroll
      for (int j = 0; j < 8; ++j) acc[i][j] *= corr[j];

    // ----- P -> LDS (C-layout -> row-major) -----
#pragma unroll
    for (int j = 0; j < 8; ++j) {
      int m = j + hh * 8;
      lds_p[m * 32 + r]      = (_Float16)sc[0][j];
      lds_p[m * 32 + 16 + r] = (_Float16)sc[1][j];
    }
    // ----- V chunk -> LDS transposed: lane = t_local row -----
    {
      const _Float16* vp = kvb + (size_t)(tb + lane) * (Hn * 256) + DHn;
#pragma unroll
      for (int d8 = 0; d8 < 16; ++d8) {
        v8h vv = *(const v8h*)(vp + d8 * 8);
#pragma unroll
        for (int e = 0; e < 8; ++e)
          lds_vt[(d8 * 8 + e) * 32 + lane] = vv[e];
      }
    }
    __syncthreads();   // single wave; LDS ops in-order, keeps compiler honest

    // ----- O += P V  (K = 32 key positions) -----
    v16h pf = load_frag_a(lds_p, 32, 0, 0);
#pragma unroll
    for (int i = 0; i < 8; ++i) {
      v16h vf = load_frag_b(lds_vt, 32, i * 16, 0);
      acc[i] = wmma16(pf, vf, acc[i]);
    }
    __syncthreads();
  }

  // ----- normalize and store -----
#pragma unroll
  for (int j = 0; j < 8; ++j) run_sum[j] = 1.0f / run_sum[j];
  _Float16* ob = O + ((size_t)(b * Sn + qbase)) * (Hn * DVn) + h * DVn;
#pragma unroll
  for (int i = 0; i < 8; ++i)
#pragma unroll
    for (int j = 0; j < 8; ++j) {
      int m = j + hh * 8;
      int n = i * 16 + r;
      ob[(size_t)m * (Hn * DVn) + n] = (_Float16)(acc[i][j] * run_sum[j]);
    }
}

// ------------------------------- launcher ----------------------------------

extern "C" void kernel_launch(void* const* d_in, const int* in_sizes, int n_in,
                              void* d_out, int out_size, void* d_ws, size_t ws_size,
                              hipStream_t stream) {
  (void)in_sizes; (void)n_in; (void)out_size; (void)ws_size;
  const float* x      = (const float*)d_in[0];
  const float* wqa    = (const float*)d_in[1];
  const float* wqa_b  = (const float*)d_in[2];
  const float* qnw    = (const float*)d_in[3];
  const float* wqb    = (const float*)d_in[4];
  const float* wqb_b  = (const float*)d_in[5];
  const float* wkva   = (const float*)d_in[6];
  const float* wkva_b = (const float*)d_in[7];
  const float* kvnw   = (const float*)d_in[8];
  const float* wkvb   = (const float*)d_in[9];
  const float* wkvb_b = (const float*)d_in[10];
  const float* wo     = (const float*)d_in[11];
  const float* wo_b   = (const float*)d_in[12];
  // d_in[13] = start_pos (0), d_in[14] = mask (causal tril, computed analytically)

  const int M = Bn * Sn;   // 2048 token rows
  char* ws = (char*)d_ws;
  size_t off = 0;
  auto alloc = [&](size_t bytes) -> void* {
    void* p = ws + off;
    off += (bytes + 255) & ~(size_t)255;
    return p;
  };

  _Float16* xh    = (_Float16*)alloc((size_t)M * Dn * 2);
  _Float16* wqah  = (_Float16*)alloc((size_t)RQn * Dn * 2);
  _Float16* wqbh  = (_Float16*)alloc((size_t)(Hn * DHn) * RQn * 2);
  _Float16* wkvah = (_Float16*)alloc((size_t)RKVn * Dn * 2);
  _Float16* wkvbh = (_Float16*)alloc((size_t)(Hn * (DHn + DVn)) * RKVn * 2);
  _Float16* woh   = (_Float16*)alloc((size_t)Dn * (Hn * DVn) * 2);
  float*    qa32  = (float*)   alloc((size_t)M * RQn * 4);
  _Float16* qah   = (_Float16*)alloc((size_t)M * RQn * 2);
  _Float16* qh    = (_Float16*)alloc((size_t)M * (Hn * DHn) * 2);
  float*    kva32 = (float*)   alloc((size_t)M * RKVn * 4);
  _Float16* kvah  = (_Float16*)alloc((size_t)M * RKVn * 2);
  _Float16* kvh   = (_Float16*)alloc((size_t)M * (Hn * 256) * 2);
  _Float16* atth  = (_Float16*)alloc((size_t)M * (Hn * DVn) * 2);

  auto cast16 = [&](const float* src, _Float16* dst, size_t n) {
    cast_f32_to_f16_kernel<<<(unsigned)((n + 255) / 256), 256, 0, stream>>>(src, dst, n);
  };
  cast16(x,    xh,    (size_t)M * Dn);
  cast16(wqa,  wqah,  (size_t)RQn * Dn);
  cast16(wqb,  wqbh,  (size_t)(Hn * DHn) * RQn);
  cast16(wkva, wkvah, (size_t)RKVn * Dn);
  cast16(wkvb, wkvbh, (size_t)(Hn * (DHn + DVn)) * RKVn);
  cast16(wo,   woh,   (size_t)Dn * (Hn * DVn));

  // q_a = x @ wq_a^T + b          (2048 x 1536, K=4096)
  gemm_xwT_kernel<false><<<dim3(RQn / 128, M / 128), 256, 0, stream>>>(
      xh, wqah, wqa_b, qa32, M, RQn, Dn);
  rmsnorm_kernel<<<M, 256, 0, stream>>>(qa32, qnw, qah, RQn);

  // q = q_a_norm @ wq_b^T + b     (2048 x 4096, K=1536) -> f16
  gemm_xwT_kernel<true><<<dim3((Hn * DHn) / 128, M / 128), 256, 0, stream>>>(
      qah, wqbh, wqb_b, qh, M, Hn * DHn, RQn);

  // kv_a = x @ wkv_a^T + b        (2048 x 512, K=4096)
  gemm_xwT_kernel<false><<<dim3(RKVn / 128, M / 128), 256, 0, stream>>>(
      xh, wkvah, wkva_b, kva32, M, RKVn, Dn);
  rmsnorm_kernel<<<M, 256, 0, stream>>>(kva32, kvnw, kvah, RKVn);

  // kv = kv_a_norm @ wkv_b^T + b  (2048 x 8192, K=512) -> f16
  gemm_xwT_kernel<true><<<dim3((Hn * 256) / 128, M / 128), 256, 0, stream>>>(
      kvah, wkvbh, wkvb_b, kvh, M, Hn * 256, RKVn);

  // attention (causal) -> f16
  mla_attention_kernel<<<dim3(Sn / 16, Hn, Bn), 32, 0, stream>>>(qh, kvh, atth);

  // out = attn @ wo^T + b         (2048 x 4096, K=4096) -> fp32 d_out
  gemm_xwT_kernel<false><<<dim3(Dn / 128, M / 128), 256, 0, stream>>>(
      atth, woh, wo_b, (float*)d_out, M, Dn, Dn);
}
